// MultiHeadAttention_30786325577824
// MI455X (gfx1250) — matrix operand order repaired and checked by reading, not compiled, not run on hardware
//
#include <hip/hip_runtime.h>

// Problem constants (B=1)
#define T_SEQ  4096
#define DMODEL 1024
#define NHEAD  16
#define HDIM   64
#define D3     (3*DMODEL)

typedef __attribute__((ext_vector_type(16))) __bf16 v16bf;
typedef __attribute__((ext_vector_type(8)))  __bf16 v8bf;
typedef __attribute__((ext_vector_type(8)))  float  v8f;
typedef __attribute__((ext_vector_type(8)))  unsigned short ush8;
typedef __attribute__((ext_vector_type(4)))  unsigned short ush4;

// -------- CDNA5 async global->LDS staging via inline asm --------------------
// global_load_async_to_lds_b128: VDST = LDS byte address VGPR, VADDR = 64-bit
// global address (GV form, saddr=off). Tracked by ASYNCcnt.
// Generic LDS pointers carry the LDS byte offset in their low 32 bits.
__device__ __forceinline__ void cp16(unsigned short* lds_dst,
                                     const unsigned short* g_src) {
    unsigned int loff = (unsigned int)(size_t)lds_dst;
    asm volatile("global_load_async_to_lds_b128 %0, %1, off"
                 :
                 : "v"(loff), "v"(g_src)
                 : "memory");
}

__device__ __forceinline__ void wait_async() {
    asm volatile("s_wait_asynccnt 0x0" ::: "memory");
}

// ---------------------------------------------------------------------------
__device__ __forceinline__ unsigned short f32_to_bf16(float f) {
    unsigned int u = __float_as_uint(f);
    unsigned int r = (u + 0x7FFFu + ((u >> 16) & 1u)) >> 16;   // RNE
    return (unsigned short)r;
}

__device__ __forceinline__ v8f zero8() {
    v8f z;
#pragma unroll
    for (int i = 0; i < 8; ++i) z[i] = 0.0f;
    return z;
}

// Build a 16-element bf16 fragment from two aligned 8-half runs in LDS.
__device__ __forceinline__ v16bf make_frag(const unsigned short* lo,
                                           const unsigned short* hi) {
    v8bf a = *(const v8bf*)lo;
    v8bf b = *(const v8bf*)hi;
    v16bf r;
#pragma unroll
    for (int i = 0; i < 8; ++i) { r[i] = a[i]; r[i + 8] = b[i]; }
    return r;
}

// ---------------------------------------------------------------------------
// Kernel 1: fp32 -> bf16 conversion (vectorized x4)
// ---------------------------------------------------------------------------
__global__ void cvt_f32_bf16(const float* __restrict__ in,
                             unsigned short* __restrict__ out, int n4) {
    int i = blockIdx.x * blockDim.x + threadIdx.x;
    if (i < n4) {
        float4 v = ((const float4*)in)[i];
        ush4 o;
        o[0] = f32_to_bf16(v.x);
        o[1] = f32_to_bf16(v.y);
        o[2] = f32_to_bf16(v.z);
        o[3] = f32_to_bf16(v.w);
        ((ush4*)out)[i] = o;
    }
}

// ---------------------------------------------------------------------------
// Kernel 2: bf16 GEMM  C[M,N] = A[M,K] * B[K,N]   (row-major, fp32 accum)
// Block = 256 threads (8 waves), tile 128x128, K-step 32, double-buffered LDS.
// Wave w owns the 32x64 sub-tile ((w>>1)*32, (w&1)*64): 2x4 frags, 8 WMMA/step.
// ---------------------------------------------------------------------------
template <bool OUT_F32>
__global__ __launch_bounds__(256)
void gemm_bf16_k(const unsigned short* __restrict__ A,
                 const unsigned short* __restrict__ B,
                 float* __restrict__ Cf, unsigned short* __restrict__ Cb,
                 int M, int N, int K) {
    __shared__ unsigned short sA[2][128][40];    // rows x k-step (16B-aligned runs)
    __shared__ unsigned short sB[2][32][136];    // k-step x cols

    const int tid  = threadIdx.x;
    const int lane = tid & 31;
    const int wid  = tid >> 5;
    const int lq   = lane & 15;
    const int kg   = lane >> 4;

    const int m0 = blockIdx.y * 128;
    const int n0 = blockIdx.x * 128;
    const int wm = (wid >> 1) * 32;
    const int wn = (wid & 1) * 64;

    v8f acc[2][4];
#pragma unroll
    for (int a = 0; a < 2; ++a)
#pragma unroll
        for (int b = 0; b < 4; ++b) acc[a][b] = zero8();

    // cooperative staging decomposition (256 threads, 16 halves each for A & B)
    const int ar = tid & 127;            // A row in tile
    const int ac = (tid >> 7) * 16;      // A col chunk (0 / 16)
    const int br = tid & 31;             // B row in tile
    const int bc = (tid >> 5) * 16;      // B col chunk (0..7)*16

    auto stage = [&](int buf, int k0) {
        const unsigned short* ga = &A[(size_t)(m0 + ar) * K + k0 + ac];
        cp16(&sA[buf][ar][ac],     ga);
        cp16(&sA[buf][ar][ac + 8], ga + 8);
        const unsigned short* gb = &B[(size_t)(k0 + br) * N + n0 + bc];
        cp16(&sB[buf][br][bc],     gb);
        cp16(&sB[buf][br][bc + 8], gb + 8);
    };

    const int nk = K >> 5;
    stage(0, 0);

    for (int ks = 0; ks < nk; ++ks) {
        const int buf = ks & 1;
        wait_async();
        __syncthreads();                        // publish buf
        if (ks + 1 < nk) stage(buf ^ 1, (ks + 1) << 5);

        v16bf afrag[2];
#pragma unroll
        for (int mt = 0; mt < 2; ++mt)
            afrag[mt] = make_frag(&sA[buf][wm + mt * 16 + lq][kg * 8],
                                  &sA[buf][wm + mt * 16 + lq][16 + kg * 8]);
        const int kk = 8 * kg;                  // B fragment k base for this lane (table order)
#pragma unroll
        for (int nt = 0; nt < 4; ++nt) {
            v16bf bfrag;
#pragma unroll
            for (int i = 0; i < 8; ++i) {
                bfrag[i]     = __builtin_bit_cast(__bf16, sB[buf][kk + i][wn + nt * 16 + lq]);
                bfrag[i + 8] = __builtin_bit_cast(__bf16, sB[buf][16 + kk + i][wn + nt * 16 + lq]);
            }
#pragma unroll
            for (int mt = 0; mt < 2; ++mt)
                acc[mt][nt] = __builtin_amdgcn_wmma_f32_16x16x32_bf16(
                    false, afrag[mt], false, bfrag,
                    (short)0, acc[mt][nt], false, false);
        }
        __syncthreads();                        // reads done before buf is restaged
    }

#pragma unroll
    for (int mt = 0; mt < 2; ++mt)
#pragma unroll
        for (int nt = 0; nt < 4; ++nt)
#pragma unroll
            for (int j = 0; j < 8; ++j) {
                int m = m0 + wm + mt * 16 + j + 8 * kg;
                int n = n0 + wn + nt * 16 + lq;
                if (OUT_F32) Cf[(size_t)m * N + n] = acc[mt][nt][j];
                else         Cb[(size_t)m * N + n] = f32_to_bf16(acc[mt][nt][j]);
            }
}

// ---------------------------------------------------------------------------
// Kernel 3: fused causal flash-attention (bf16 QKV, fp32 softmax/accum)
// grid = (T/64, H); block = 128 threads (4 waves, 16 query rows each)
// qkv layout: row t of 3072: [Q | K | V], head h at cols h*64..h*64+64
// ---------------------------------------------------------------------------
__global__ __launch_bounds__(128)
void attn_kernel(const unsigned short* __restrict__ qkv,
                 unsigned short* __restrict__ attn_out) {
    __shared__ unsigned short sQ [64][72];        // [q-row][hd]
    __shared__ unsigned short sKt[64][72];        // [hd][key]   (transposed)
    __shared__ unsigned short sV [64][72];        // [key][hd]
    __shared__ unsigned short sP [4][16][72];     // per-wave probs, [row][key]

    const int qb   = blockIdx.x;
    const int h    = blockIdx.y;
    const int tid  = threadIdx.x;
    const int lane = tid & 31;
    const int wid  = tid >> 5;
    const int lq   = lane & 15;
    const int kgr  = lane >> 4;
    const int qr0  = wid * 16;           // wave's first query row in tile
    const float scale = 0.125f;          // 1/sqrt(64)

    // stage Q tile (64x64) — async, published by first in-loop barrier pair
    {
        int r = tid & 63, half = tid >> 6;
        const unsigned short* src =
            qkv + (size_t)(qb * 64 + r) * D3 + h * HDIM + half * 32;
#pragma unroll
        for (int i = 0; i < 4; ++i)
            cp16(&sQ[r][half * 32 + i * 8], &src[i * 8]);
    }

    v8f   acc_o[4];
    float mrun[8], lrun[8];
#pragma unroll
    for (int nt = 0; nt < 4; ++nt) acc_o[nt] = zero8();
#pragma unroll
    for (int j = 0; j < 8; ++j) { mrun[j] = -1e30f; lrun[j] = 0.0f; }

    for (int kb = 0; kb <= qb; ++kb) {
        __syncthreads();                  // previous reads of sKt/sV complete
        {
            int key = tid & 63, half = tid >> 6;
            // K tile transposed (register-mediated scatter)
            const unsigned short* ksrc =
                qkv + (size_t)(kb * 64 + key) * D3 + DMODEL + h * HDIM + half * 32;
            ush8 tmp[4];
#pragma unroll
            for (int i = 0; i < 4; ++i) tmp[i] = *(const ush8*)&ksrc[i * 8];
#pragma unroll
            for (int i = 0; i < 4; ++i)
#pragma unroll
                for (int e = 0; e < 8; ++e)
                    sKt[half * 32 + i * 8 + e][key] = tmp[i][e];

            // V tile (async direct-to-LDS)
            const unsigned short* vsrc =
                qkv + (size_t)(kb * 64 + key) * D3 + 2 * DMODEL + h * HDIM + half * 32;
#pragma unroll
            for (int i = 0; i < 4; ++i)
                cp16(&sV[key][half * 32 + i * 8], &vsrc[i * 8]);
        }
        wait_async();
        __syncthreads();                  // publish sKt/sV (and sQ on iter 0)

        // S = Q * K^T  (wave: 16 q-rows x 64 keys)
        v8f acc_s[4];
#pragma unroll
        for (int nt = 0; nt < 4; ++nt) acc_s[nt] = zero8();
#pragma unroll
        for (int ks = 0; ks < 2; ++ks) {
            v16bf a = make_frag(&sQ[qr0 + lq][ks * 32 + kgr * 8],
                                &sQ[qr0 + lq][ks * 32 + 16 + kgr * 8]);
            const int hd = ks * 32 + 8 * kgr;   // B fragment k base (table order)
#pragma unroll
            for (int nt = 0; nt < 4; ++nt) {
                v16bf b;
#pragma unroll
                for (int i = 0; i < 8; ++i) {
                    b[i]     = __builtin_bit_cast(__bf16, sKt[hd + i][nt * 16 + lq]);
                    b[i + 8] = __builtin_bit_cast(__bf16, sKt[hd + 16 + i][nt * 16 + lq]);
                }
                acc_s[nt] = __builtin_amdgcn_wmma_f32_16x16x32_bf16(
                    false, a, false, b, (short)0, acc_s[nt], false, false);
            }
        }

        // scale + causal mask
        float sreg[4][8];
        const bool diag = (kb == qb);
#pragma unroll
        for (int nt = 0; nt < 4; ++nt)
#pragma unroll
            for (int j = 0; j < 8; ++j) {
                float s = acc_s[nt][j] * scale;
                if (diag) {
                    int qg = qb * 64 + qr0 + j + 8 * kgr;
                    int kk = kb * 64 + nt * 16 + lq;
                    if (kk > qg) s = -1e30f;
                }
                sreg[nt][j] = s;
            }

        // online softmax per row (16-lane shuffle reductions stay in-group)
#pragma unroll
        for (int j = 0; j < 8; ++j) {
            float mx = sreg[0][j];
#pragma unroll
            for (int nt = 1; nt < 4; ++nt) mx = fmaxf(mx, sreg[nt][j]);
#pragma unroll
            for (int off = 1; off < 16; off <<= 1)
                mx = fmaxf(mx, __shfl_xor(mx, off, 32));

            float mnew = fmaxf(mrun[j], mx);
            float corr = __expf(mrun[j] - mnew);
            float rs = 0.0f;
#pragma unroll
            for (int nt = 0; nt < 4; ++nt) {
                float p = __expf(sreg[nt][j] - mnew);
                rs += p;
                sP[wid][j + 8 * kgr][nt * 16 + lq] = f32_to_bf16(p);
            }
#pragma unroll
            for (int off = 1; off < 16; off <<= 1)
                rs += __shfl_xor(rs, off, 32);

            lrun[j] = lrun[j] * corr + rs;
            mrun[j] = mnew;
#pragma unroll
            for (int nt = 0; nt < 4; ++nt) acc_o[nt][j] *= corr;
        }

        // O += P * V   (A frags from sP, B frags from sV)
#pragma unroll
        for (int ks = 0; ks < 2; ++ks) {
            v16bf a = make_frag(&sP[wid][lq][ks * 32 + kgr * 8],
                                &sP[wid][lq][ks * 32 + 16 + kgr * 8]);
            const int kv = ks * 32 + 8 * kgr;   // V key base for this lane (table order)
#pragma unroll
            for (int nt = 0; nt < 4; ++nt) {
                v16bf b;
#pragma unroll
                for (int i = 0; i < 8; ++i) {
                    b[i]     = __builtin_bit_cast(__bf16, sV[kv + i][nt * 16 + lq]);
                    b[i + 8] = __builtin_bit_cast(__bf16, sV[kv + 16 + i][nt * 16 + lq]);
                }
                acc_o[nt] = __builtin_amdgcn_wmma_f32_16x16x32_bf16(
                    false, a, false, b, (short)0, acc_o[nt], false, false);
            }
        }
    }

    // normalize and write out (bf16, layout [T][D] with head offset)
#pragma unroll
    for (int j = 0; j < 8; ++j) {
        float il = 1.0f / lrun[j];
        int row = qb * 64 + qr0 + j + 8 * kgr;
#pragma unroll
        for (int nt = 0; nt < 4; ++nt) {
            int col = h * HDIM + nt * 16 + lq;
            attn_out[(size_t)row * DMODEL + col] = f32_to_bf16(acc_o[nt][j] * il);
        }
    }
}

// ---------------------------------------------------------------------------
// Host launcher
// ---------------------------------------------------------------------------
extern "C" void kernel_launch(void* const* d_in, const int* in_sizes, int n_in,
                              void* d_out, int out_size, void* d_ws, size_t ws_size,
                              hipStream_t stream) {
    const float* x     = (const float*)d_in[0];
    const float* w_qkv = (const float*)d_in[1];
    const float* w_out = (const float*)d_in[2];
    float* out = (float*)d_out;

    char* ws = (char*)d_ws;
    const size_t MB = 1024 * 1024;
    unsigned short* xb   = (unsigned short*)(ws + 0 * MB);   //  8 MB: bf16 x
    unsigned short* wqb  = (unsigned short*)(ws + 8 * MB);   //  6 MB: bf16 w_qkv
    unsigned short* wob  = (unsigned short*)(ws + 14 * MB);  //  2 MB: bf16 w_out
    unsigned short* qkvb = (unsigned short*)(ws + 16 * MB);  // 24 MB: bf16 QKV
    unsigned short* attb = (unsigned short*)(ws + 40 * MB);  //  8 MB: bf16 attn out

    {   // conversions
        int n4 = (T_SEQ * DMODEL) / 4;
        cvt_f32_bf16<<<(n4 + 255) / 256, 256, 0, stream>>>(x, xb, n4);
        n4 = (DMODEL * D3) / 4;
        cvt_f32_bf16<<<(n4 + 255) / 256, 256, 0, stream>>>(w_qkv, wqb, n4);
        n4 = (DMODEL * DMODEL) / 4;
        cvt_f32_bf16<<<(n4 + 255) / 256, 256, 0, stream>>>(w_out, wob, n4);
    }

    // QKV projection: (4096x1024)x(1024x3072) -> bf16
    dim3 g1(D3 / 128, T_SEQ / 128);
    gemm_bf16_k<false><<<g1, 256, 0, stream>>>(xb, wqb, nullptr, qkvb,
                                               T_SEQ, D3, DMODEL);

    // fused causal attention
    dim3 g2(T_SEQ / 64, NHEAD);
    attn_kernel<<<g2, 128, 0, stream>>>(qkvb, attb);

    // output projection: (4096x1024)x(1024x1024) -> fp32
    dim3 g3(DMODEL / 128, T_SEQ / 128);
    gemm_bf16_k<true><<<g3, 256, 0, stream>>>(attb, wob, out, nullptr,
                                              T_SEQ, DMODEL, DMODEL);
}
